// MambaDecoder_87162066305523
// MI455X (gfx1250) — compile-verified, matrix operand order checked by reading, not run on hardware
//
#include <hip/hip_runtime.h>
#include <hip/hip_bf16.h>

// ---------------------------------------------------------------------------
// MambaDecoder on MI455X (gfx1250): bf16 WMMA GEMMs with async-to-LDS weight
// staging + fp32 register scan.
// B=8, L=512, DM=384, D_IN=768, N_ST=16, DT_R=24, K=4, DEPTH=4, R_TOK=256
// ---------------------------------------------------------------------------

typedef __attribute__((ext_vector_type(16))) __bf16 v16bf;
typedef __attribute__((ext_vector_type(8)))  float  v8f;

#define Bc   8
#define Lc   512
#define DMc  384
#define DINc 768
#define NSTc 16
#define DTRc 24
#define MTc  (Bc * Lc)   // 4096 tokens

#define LDS_PAD 8        // bf16 elements of per-row pad: (K+8)*2 = 97*16B -> bank rotation

// -------------------------------------------------------------------- utils
__global__ void cvt_bf16_kernel(const float* __restrict__ in,
                                __hip_bfloat16* __restrict__ out, int n) {
    int i = blockIdx.x * 256 + threadIdx.x;
    if (i < n) out[i] = __float2bfloat16(in[i]);
}

// x_proj weights are (56, 768); pad rows to 64 (zeros) for the WMMA GEMM.
__global__ void pad_xw_bf16_kernel(const float* __restrict__ xw,
                                   __hip_bfloat16* __restrict__ out) {
    int i = blockIdx.x * 256 + threadIdx.x;   // < 64*768
    int j = i / DINc;
    int r = i - j * DINc;
    float v = (j < (DTRc + 2 * NSTc)) ? xw[j * DINc + r] : 0.0f;
    out[i] = __float2bfloat16(v);
}

// ------------------------------------------- fused residual-add + LayerNorm
__global__ void prelude_ln_kernel(const float* __restrict__ hin,
                                  const float* __restrict__ pos,
                                  const float* __restrict__ w,
                                  const float* __restrict__ b,
                                  float* __restrict__ hp,
                                  __hip_bfloat16* __restrict__ hn) {
    int tk = blockIdx.x;
    int tid = threadIdx.x;
    __shared__ float red[128];
    const float* pi = hin + (size_t)tk * DMc;
    const float* pp = pos + (size_t)tk * DMc;
    float v0 = pi[tid]       + pp[tid];
    float v1 = pi[tid + 128] + pp[tid + 128];
    float v2 = pi[tid + 256] + pp[tid + 256];
    float* ph = hp + (size_t)tk * DMc;
    ph[tid] = v0; ph[tid + 128] = v1; ph[tid + 256] = v2;

    red[tid] = v0 + v1 + v2;
    __syncthreads();
    for (int s = 64; s > 0; s >>= 1) {
        if (tid < s) red[tid] += red[tid + s];
        __syncthreads();
    }
    float mu = red[0] * (1.0f / DMc);
    __syncthreads();
    float d0 = v0 - mu, d1 = v1 - mu, d2 = v2 - mu;
    red[tid] = d0 * d0 + d1 * d1 + d2 * d2;
    __syncthreads();
    for (int s = 64; s > 0; s >>= 1) {
        if (tid < s) red[tid] += red[tid + s];
        __syncthreads();
    }
    float rstd = rsqrtf(red[0] * (1.0f / DMc) + 1e-5f);
    __hip_bfloat16* pn = hn + (size_t)tk * DMc;
    pn[tid]       = __float2bfloat16(d0 * rstd * w[tid]       + b[tid]);
    pn[tid + 128] = __float2bfloat16(d1 * rstd * w[tid + 128] + b[tid + 128]);
    pn[tid + 256] = __float2bfloat16(d2 * rstd * w[tid + 256] + b[tid + 256]);
}

// ------------------------------------------------------------ WMMA GEMM
// C[M,N](f32) = A[M,K](bf16) * W[N,K](bf16)^T  (+ resid if RESID)
// Block: 256 threads = 8 waves; macro tile 128(M) x 64(N).
// The 64xK weight tile is staged ONCE per block into LDS with
// GLOBAL_LOAD_ASYNC_TO_LDS_B128 (ASYNCcnt), then every wave feeds its B
// fragments from LDS (ds_load_b128, padded stride -> conflict-free), removing
// the per-WMMA global loadcnt stall and the 8x intra-block W overfetch.
// A fragments stream from global/L2 with prefetch (row-private per wave).
template <int RESID>
__global__ void wmma_gemm_kernel(const __hip_bfloat16* __restrict__ A,
                                 const __hip_bfloat16* __restrict__ W,
                                 const float* __restrict__ resid,
                                 float* __restrict__ C,
                                 int M, int N, int K) {
    __shared__ __hip_bfloat16 ldsW[64 * (768 + LDS_PAD)];   // 97 KB, worst-case K

    int wave = threadIdx.x >> 5;
    int lane = threadIdx.x & 31;
    int row  = lane & 15;     // M (for A) / N (for B) index within tile
    int kh   = lane >> 4;     // K half select
    int m0 = blockIdx.x * 128 + wave * 16;
    int n0 = blockIdx.y * 64;
    int ldsStride = K + LDS_PAD;

    // ---- async stage: 64 rows x K bf16 of W into LDS (16B chunks per lane)
    {
        unsigned lbase = (unsigned)(size_t)(void*)&ldsW[0];
        int cpr   = (K * 2) / 16;          // 16B chunks per row (48 or 96)
        int total = 64 * cpr;
        for (int c = threadIdx.x; c < total; c += 256) {
            int r  = c / cpr;
            int cc = c - r * cpr;
            const __hip_bfloat16* gsrc = W + (size_t)(n0 + r) * K + cc * 8;
            unsigned ldst = lbase + (unsigned)((r * ldsStride + cc * 8) * 2);
            asm volatile("global_load_async_to_lds_b128 %0, %1, off"
                         :: "v"(ldst), "v"(gsrc) : "memory");
        }
#if defined(__has_builtin)
#if __has_builtin(__builtin_amdgcn_s_wait_asynccnt)
        __builtin_amdgcn_s_wait_asynccnt(0);
#else
        asm volatile("s_wait_asynccnt 0x0" ::: "memory");
#endif
#else
        asm volatile("s_wait_asynccnt 0x0" ::: "memory");
#endif
        __syncthreads();
    }

    v8f acc0 = {}, acc1 = {}, acc2 = {}, acc3 = {};
    const __hip_bfloat16* arow = A + (size_t)(m0 + row) * K + kh * 16;
    const __hip_bfloat16* l0 = ldsW + (size_t)(row)      * ldsStride + kh * 16;
    const __hip_bfloat16* l1 = ldsW + (size_t)(16 + row) * ldsStride + kh * 16;
    const __hip_bfloat16* l2 = ldsW + (size_t)(32 + row) * ldsStride + kh * 16;
    const __hip_bfloat16* l3 = ldsW + (size_t)(48 + row) * ldsStride + kh * 16;

    for (int k0 = 0; k0 < K; k0 += 32) {
        v16bf a  = *(const v16bf*)(arow + k0);
        v16bf b0 = *(const v16bf*)(l0 + k0);
        v16bf b1 = *(const v16bf*)(l1 + k0);
        v16bf b2 = *(const v16bf*)(l2 + k0);
        v16bf b3 = *(const v16bf*)(l3 + k0);
        if (k0 + 32 < K) __builtin_prefetch(arow + k0 + 32, 0, 3);
        acc0 = __builtin_amdgcn_wmma_f32_16x16x32_bf16(false, a, false, b0,
                                                       (short)0, acc0, false, false);
        acc1 = __builtin_amdgcn_wmma_f32_16x16x32_bf16(false, a, false, b1,
                                                       (short)0, acc1, false, false);
        acc2 = __builtin_amdgcn_wmma_f32_16x16x32_bf16(false, a, false, b2,
                                                       (short)0, acc2, false, false);
        acc3 = __builtin_amdgcn_wmma_f32_16x16x32_bf16(false, a, false, b3,
                                                       (short)0, acc3, false, false);
    }

    // C/D layout: lanes 0-15 -> M=r, lanes 16-31 -> M=r+8; N = lane&15.
    int col = row;
    int rb  = kh * 8;
#pragma unroll
    for (int r = 0; r < 8; r++) {
        int mrow = m0 + rb + r;
        size_t base = (size_t)mrow * N + n0 + col;
        float o0 = acc0[r], o1 = acc1[r], o2 = acc2[r], o3 = acc3[r];
        if (RESID) {
            o0 += resid[base];      o1 += resid[base + 16];
            o2 += resid[base + 32]; o3 += resid[base + 48];
        }
        C[base]      = o0; C[base + 16] = o1;
        C[base + 32] = o2; C[base + 48] = o3;
    }
}

// ------------------------------------------------- causal conv1d(K=4) + SiLU
__global__ void conv_silu_kernel(const float* __restrict__ xz,
                                 const float* __restrict__ cw,
                                 const float* __restrict__ cb,
                                 float* __restrict__ xc, int dir) {
    int idx = blockIdx.x * 256 + threadIdx.x;   // < MT*DIN
    int d = idx % DINc;
    int t = idx / DINc;
    int l = t % Lc;
    int b = t / Lc;
    float acc = cb[d];
#pragma unroll
    for (int k = 0; k < 4; k++) {
        int ls = l - 3 + k;
        if (ls >= 0) {
            int lo = dir ? (Lc - 1 - ls) : ls;
            acc += xz[(size_t)(b * Lc + lo) * (2 * DINc) + d] * cw[d * 4 + k];
        }
    }
    xc[idx] = acc / (1.0f + __expf(-acc));     // silu
}

// ------------------------------------------------------------ selective scan
__global__ void scan_dir_kernel(const float* __restrict__ dbl,
                                const float* __restrict__ xc,
                                const float* __restrict__ xz,
                                const float* __restrict__ dw,
                                const float* __restrict__ dbias,
                                const float* __restrict__ alog,
                                const float* __restrict__ Dvec,
                                float* __restrict__ ysum,
                                int dir, int accum) {
    int t = blockIdx.x * 256 + threadIdx.x;   // < B*DIN
    int d = t % DINc;
    int b = t / DINc;

    float dwr[DTRc];
#pragma unroll
    for (int r = 0; r < DTRc; r++) dwr[r] = dw[d * DTRc + r];
    float bias = dbias[d];
    float Ad[NSTc];
#pragma unroll
    for (int n = 0; n < NSTc; n++) Ad[n] = -__expf(alog[d * NSTc + n]);
    float Dd = Dvec[d];
    float h[NSTc];
#pragma unroll
    for (int n = 0; n < NSTc; n++) h[n] = 0.0f;

    for (int l = 0; l < Lc; l++) {
        const float* row = dbl + (size_t)(b * Lc + l) * 64;
        float dt = bias;
#pragma unroll
        for (int r = 0; r < DTRc; r++) dt += row[r] * dwr[r];
        dt = (dt > 20.0f) ? dt : log1pf(__expf(dt));          // softplus
        float xcv = xc[(size_t)(b * Lc + l) * DINc + d];
        float y = 0.0f;
#pragma unroll
        for (int n = 0; n < NSTc; n++) {
            float dA = __expf(dt * Ad[n]);
            h[n] = dA * h[n] + dt * row[DTRc + n] * xcv;      // B block
            y += h[n] * row[DTRc + NSTc + n];                 // C block
        }
        y += xcv * Dd;
        int lo = dir ? (Lc - 1 - l) : l;
        float z = xz[(size_t)(b * Lc + lo) * (2 * DINc) + DINc + d];
        float g = z / (1.0f + __expf(-z));                    // silu gate
        float outv = y * g;
        size_t oi = (size_t)(b * Lc + lo) * DINc + d;
        if (accum) ysum[oi] += outv; else ysum[oi] = outv;
    }
}

// ---------------------------------------------------------- final LayerNorm
__global__ void final_ln_kernel(const float* __restrict__ h,
                                const float* __restrict__ w,
                                const float* __restrict__ b,
                                float* __restrict__ out) {
    int tk = blockIdx.x;              // < B*256
    int tid = threadIdx.x;
    int bi = tk >> 8;
    int l = (Lc - 256) + (tk & 255);
    __shared__ float red[128];
    const float* pi = h + (size_t)(bi * Lc + l) * DMc;
    float v0 = pi[tid], v1 = pi[tid + 128], v2 = pi[tid + 256];
    red[tid] = v0 + v1 + v2;
    __syncthreads();
    for (int s = 64; s > 0; s >>= 1) {
        if (tid < s) red[tid] += red[tid + s];
        __syncthreads();
    }
    float mu = red[0] * (1.0f / DMc);
    __syncthreads();
    float d0 = v0 - mu, d1 = v1 - mu, d2 = v2 - mu;
    red[tid] = d0 * d0 + d1 * d1 + d2 * d2;
    __syncthreads();
    for (int s = 64; s > 0; s >>= 1) {
        if (tid < s) red[tid] += red[tid + s];
        __syncthreads();
    }
    float rstd = rsqrtf(red[0] * (1.0f / DMc) + 1e-5f);
    float* po = out + (size_t)tk * DMc;
    po[tid]       = d0 * rstd * w[tid]       + b[tid];
    po[tid + 128] = d1 * rstd * w[tid + 128] + b[tid + 128];
    po[tid + 256] = d2 * rstd * w[tid + 256] + b[tid + 256];
}

// ---------------------------------------------------------------- launcher
extern "C" void kernel_launch(void* const* d_in, const int* in_sizes, int n_in,
                              void* d_out, int out_size, void* d_ws, size_t ws_size,
                              hipStream_t stream) {
    (void)in_sizes; (void)n_in; (void)out_size; (void)ws_size;

    const float* x          = (const float*)d_in[1];
    const float* pos        = (const float*)d_in[2];
    const float* norm_w     = (const float*)d_in[4];
    const float* norm_b     = (const float*)d_in[5];
    const float* in_proj_w  = (const float*)d_in[6];
    const float* conv_w     = (const float*)d_in[7];
    const float* conv_b     = (const float*)d_in[8];
    const float* xproj_w    = (const float*)d_in[9];
    const float* dtproj_w   = (const float*)d_in[10];
    const float* dtproj_b   = (const float*)d_in[11];
    const float* A_log      = (const float*)d_in[12];
    const float* Dvec       = (const float*)d_in[13];
    const float* conv_w_b   = (const float*)d_in[14];
    const float* conv_b_b   = (const float*)d_in[15];
    const float* xproj_w_b  = (const float*)d_in[16];
    const float* dtproj_w_b = (const float*)d_in[17];
    const float* dtproj_b_b = (const float*)d_in[18];
    const float* A_b_log    = (const float*)d_in[19];
    const float* D_b        = (const float*)d_in[20];
    const float* outproj_w  = (const float*)d_in[21];
    const float* fnorm_w    = (const float*)d_in[22];
    const float* fnorm_b    = (const float*)d_in[23];

    // ---- workspace carve-up (256B aligned slices)
    char* base = (char*)d_ws;
    size_t off = 0;
    auto alloc = [&](size_t bytes) -> char* {
        char* p = base + off;
        off = (off + bytes + 255) & ~(size_t)255;
        return p;
    };
    float* hp    = (float*)alloc((size_t)MTc * DMc * 4);
    float* hcur  = (float*)alloc((size_t)MTc * DMc * 4);
    float* xzb   = (float*)alloc((size_t)MTc * 2 * DINc * 4);
    float* xcb   = (float*)alloc((size_t)MTc * DINc * 4);
    float* ysum  = (float*)alloc((size_t)MTc * DINc * 4);
    float* dblb  = (float*)alloc((size_t)MTc * 64 * 4);
    __hip_bfloat16* hnbf   = (__hip_bfloat16*)alloc((size_t)MTc * DMc * 2);
    __hip_bfloat16* winbf  = (__hip_bfloat16*)alloc((size_t)2 * DINc * DMc * 2);
    __hip_bfloat16* woutbf = (__hip_bfloat16*)alloc((size_t)DMc * DINc * 2);
    __hip_bfloat16* xwbf   = (__hip_bfloat16*)alloc((size_t)64 * DINc * 2);
    __hip_bfloat16* xybf   = (__hip_bfloat16*)alloc((size_t)MTc * DINc * 2); // xc_bf16 / y_bf16

    const int n_in_w  = 2 * DINc * DMc;   // 589824
    const int n_out_w = DMc * DINc;       // 294912
    const int n_act   = MTc * DINc;       // 3145728

    for (int i = 0; i < 4; i++) {
        const float* hin = (i == 0) ? x : hcur;
        const float* inw  = in_proj_w + (size_t)i * n_in_w;
        const float* outw = outproj_w + (size_t)i * n_out_w;

        cvt_bf16_kernel<<<(n_in_w + 255) / 256, 256, 0, stream>>>(inw, winbf, n_in_w);
        cvt_bf16_kernel<<<(n_out_w + 255) / 256, 256, 0, stream>>>(outw, woutbf, n_out_w);

        prelude_ln_kernel<<<MTc, 128, 0, stream>>>(
            hin, pos, norm_w + i * DMc, norm_b + i * DMc, hp, hnbf);

        // xz = hn @ in_proj^T : (4096 x 1536), K=384
        wmma_gemm_kernel<0><<<dim3(MTc / 128, (2 * DINc) / 64), 256, 0, stream>>>(
            hnbf, winbf, nullptr, xzb, MTc, 2 * DINc, DMc);

        for (int dir = 0; dir < 2; dir++) {
            const float* cw  = (dir ? conv_w_b   : conv_w)   + (size_t)i * DINc * 4;
            const float* cb  = (dir ? conv_b_b   : conv_b)   + (size_t)i * DINc;
            const float* xw  = (dir ? xproj_w_b  : xproj_w)  + (size_t)i * 56 * DINc;
            const float* dtw = (dir ? dtproj_w_b : dtproj_w) + (size_t)i * DINc * DTRc;
            const float* dtb = (dir ? dtproj_b_b : dtproj_b) + (size_t)i * DINc;
            const float* al  = (dir ? A_b_log    : A_log)    + (size_t)i * DINc * NSTc;
            const float* dd  = (dir ? D_b        : Dvec)     + (size_t)i * DINc;

            conv_silu_kernel<<<n_act / 256, 256, 0, stream>>>(xzb, cw, cb, xcb, dir);
            cvt_bf16_kernel<<<n_act / 256, 256, 0, stream>>>(xcb, xybf, n_act);
            pad_xw_bf16_kernel<<<(64 * DINc) / 256, 256, 0, stream>>>(xw, xwbf);

            // dbl = xc @ xw^T : (4096 x 64pad), K=768
            wmma_gemm_kernel<0><<<dim3(MTc / 128, 1), 256, 0, stream>>>(
                xybf, xwbf, nullptr, dblb, MTc, 64, DINc);

            scan_dir_kernel<<<(Bc * DINc) / 256, 256, 0, stream>>>(
                dblb, xcb, xzb, dtw, dtb, al, dd, ysum, dir, /*accum=*/dir);
        }

        cvt_bf16_kernel<<<n_act / 256, 256, 0, stream>>>(ysum, xybf, n_act);

        // h = hp + y @ outproj^T : (4096 x 384), K=768, fused residual
        wmma_gemm_kernel<1><<<dim3(MTc / 128, DMc / 64), 256, 0, stream>>>(
            xybf, woutbf, hp, hcur, MTc, DMc, DINc);
    }

    final_ln_kernel<<<Bc * 256, 128, 0, stream>>>(hcur, fnorm_w, fnorm_b, (float*)d_out);
}
